// RotarySelfAttention_65206193488085
// MI455X (gfx1250) — matrix operand order repaired and checked by reading, not compiled, not run on hardware
//
#include <hip/hip_runtime.h>
#include <hip/hip_bf16.h>
#include <math.h>

typedef __attribute__((ext_vector_type(16))) _Float16 v16h;
typedef __attribute__((ext_vector_type(8)))  _Float16 v8h;
typedef __attribute__((ext_vector_type(8)))  float    v8f;
typedef __attribute__((ext_vector_type(4)))  unsigned int u32x4;
typedef __attribute__((ext_vector_type(8)))  int i32x8;
typedef __attribute__((ext_vector_type(4)))  int i32x4;

#define B_  4
#define N_  2048
#define D_  512
#define H_  8
#define DH_ 64
#define TD_ 1536   // 3*D

// LDS tile geometry (halfs). Row strides padded to 16B multiples and odd dword
// groups so 16-lane fragment reads hit distinct banks.
#define KSTR 72    // 64 + 8  (36 dwords per row)
#define VSTR 136   // 128 + 8 (68 dwords per row)
#define PSTR 136
#define KTILE_H (128 * KSTR)   // 9216 halfs per buffer
#define VTILE_H (64 * VSTR)    // 8704 halfs per buffer
#define PTILE_H (8 * 16 * PSTR)
#define SMEM_BYTES ((2 * KTILE_H + 2 * VTILE_H + PTILE_H) * 2)  // 106496 B

#if defined(__AMDGCN__) && __has_builtin(__builtin_amdgcn_tensor_load_to_lds) && \
    __has_builtin(__builtin_amdgcn_s_wait_tensorcnt)
#define USE_TDM 1
#else
#define USE_TDM 0
#endif

__device__ __forceinline__ v8f wmma_f16f32(v16h a, v16h b, v8f c) {
  return __builtin_amdgcn_wmma_f32_16x16x32_f16(
      false, a, false, b, (short)0, c, false, false);
}

__device__ __forceinline__ v16h cat8(v8h lo, v8h hi) {
  return __builtin_shufflevector(lo, hi, 0,1,2,3,4,5,6,7,8,9,10,11,12,13,14,15);
}

#if USE_TDM
// Issue one 2D TDM tile load: global (row-major, stride0 elements between rows)
// -> LDS with per-row padding (pad codes per D# group1 spec).
__device__ __forceinline__ void tdm_load_2d(unsigned lds_addr, const void* gaddr,
                                            unsigned tile_d0, unsigned tile_d1,
                                            unsigned tensor_d0, unsigned tensor_d1,
                                            unsigned long long stride0,
                                            unsigned pad_interval_code,
                                            unsigned pad_amount_code) {
  const unsigned long long ga = (unsigned long long)(uintptr_t)gaddr;
  u32x4 g0;
  g0[0] = 1u;                                        // count=1, user mode
  g0[1] = lds_addr;                                  // lds_addr (bytes)
  g0[2] = (unsigned)(ga & 0xffffffffu);              // global_addr[31:0]
  g0[3] = (unsigned)((ga >> 32) & 0x01ffffffu)       // global_addr[56:32]
          | (2u << 30);                              // type = 2 (image)
  i32x8 g1;
  g1[0] = (int)((1u << 16)                           // data_size = 2 bytes
                | (1u << 20)                         // pad_enable
                | (pad_interval_code << 22) | (pad_amount_code << 25));
  g1[1] = (int)((tensor_d0 & 0xffffu) << 16);        // tensor_dim0[15:0]
  g1[2] = (int)(((tensor_d0 >> 16) & 0xffffu) | ((tensor_d1 & 0xffffu) << 16));
  g1[3] = (int)(((tensor_d1 >> 16) & 0xffffu) | ((tile_d0 & 0xffffu) << 16));
  g1[4] = (int)(tile_d1 & 0xffffu);                  // tile_dim1 (tile_dim2=0)
  g1[5] = (int)(stride0 & 0xffffffffu);              // tensor_dim0_stride[31:0]
  g1[6] = (int)((stride0 >> 32) & 0xffffu);          // [47:32]; dim1_stride=0
  g1[7] = 0;
  i32x4 z4 = {0, 0, 0, 0};
#if __clang_major__ >= 23
  i32x8 z8 = {0, 0, 0, 0, 0, 0, 0, 0};
  __builtin_amdgcn_tensor_load_to_lds(g0, g1, z4, z4, z8, 0);
#else
  __builtin_amdgcn_tensor_load_to_lds(g0, g1, z4, z4, 0);
#endif
}
#endif

// ---------------------------------------------------------------- casts
__global__ void cast_f32_to_f16(const float* __restrict__ src,
                                _Float16* __restrict__ dst, int n) {
  for (int i = blockIdx.x * blockDim.x + threadIdx.x; i < n;
       i += gridDim.x * blockDim.x)
    dst[i] = (_Float16)src[i];
}

// ---------------------------------------------------------------- GEMM: C = A(MxK) * W(NOUTxK)^T + bias
template<int NOUT, bool OUT_HALF>
__global__ void __launch_bounds__(256)
gemm_xwT(const _Float16* __restrict__ A, const _Float16* __restrict__ W,
         const float* __restrict__ bias, void* __restrict__ Cout, int K) {
  const int lane = threadIdx.x & 31;
  const int wave = threadIdx.x >> 5;
  const int lhi  = lane >> 4;
  const int lsub = lane & 15;
  const int m0 = blockIdx.x * 128 + wave * 16;
  const int n0 = blockIdx.y * 64;
  const size_t arow = (size_t)(m0 + lsub) * K;

  v8f acc[4] = {};
#pragma unroll 2
  for (int kk = 0; kk < K; kk += 32) {
    __builtin_prefetch(A + arow + kk + 128, 0, 1);
    v16h af = *(const v16h*)(A + arow + kk + 16 * lhi);
    v16h bf[4];
#pragma unroll
    for (int nc = 0; nc < 4; ++nc)
      bf[nc] = *(const v16h*)(W + (size_t)(n0 + nc * 16 + lsub) * K + kk + 16 * lhi);
#pragma unroll
    for (int nc = 0; nc < 4; ++nc)
      acc[nc] = wmma_f16f32(af, bf[nc], acc[nc]);
  }
#pragma unroll
  for (int nc = 0; nc < 4; ++nc) {
    const int col = n0 + nc * 16 + lsub;
    const float bv = bias[col];
#pragma unroll
    for (int r = 0; r < 8; ++r) {
      const int row = m0 + r + 8 * lhi;
      const float v = acc[nc][r] + bv;
      if (OUT_HALF) ((_Float16*)Cout)[(size_t)row * NOUT + col] = (_Float16)v;
      else          ((float*)Cout)[(size_t)row * NOUT + col] = v;
    }
  }
}

// ---------------------------------------------------------------- rotary + head split
__global__ void rot_split(const _Float16* __restrict__ qkv, const float* __restrict__ pos,
                          _Float16* __restrict__ Qh, _Float16* __restrict__ Kh,
                          _Float16* __restrict__ Vt) {
  const int idx = blockIdx.x * blockDim.x + threadIdx.x;   // (b*N + n)*H + h
  if (idx >= B_ * N_ * H_) return;
  const int h = idx & (H_ - 1);
  const int t = idx >> 3;
  const int n = t & (N_ - 1);
  const int b = t >> 11;

  float sn[3], cs[3];
#pragma unroll
  for (int i = 0; i < 3; ++i) {
    const float th = pos[((size_t)(b * N_ + n)) * 3 + i] * 3.14159265358979323846f;
    sn[i] = __sinf(th);
    cs[i] = __cosf(th);
  }
  const _Float16* row = qkv + (size_t)(b * N_ + n) * TD_;
  const _Float16* qp = row + h * DH_;
  const _Float16* kp = row + D_ + h * DH_;
  const _Float16* vp = row + 2 * D_ + h * DH_;
  const size_t obase = ((size_t)(b * H_ + h) * N_ + n) * DH_;

#pragma unroll
  for (int i = 0; i < 3; ++i) {
    const float q1 = (float)qp[2 * i], q2 = (float)qp[2 * i + 1];
    Qh[obase + i]     = (_Float16)(q1 * cs[i] - q2 * sn[i]);
    Qh[obase + 3 + i] = (_Float16)(q1 * sn[i] + q2 * cs[i]);
    const float k1 = (float)kp[2 * i], k2 = (float)kp[2 * i + 1];
    Kh[obase + i]     = (_Float16)(k1 * cs[i] - k2 * sn[i]);
    Kh[obase + 3 + i] = (_Float16)(k1 * sn[i] + k2 * cs[i]);
  }
  for (int d = 6; d < DH_; ++d) { Qh[obase + d] = qp[d]; Kh[obase + d] = kp[d]; }

  const size_t vtb = (size_t)(b * H_ + h) * DH_ * N_;
  for (int d = 0; d < DH_; ++d) Vt[vtb + (size_t)d * N_ + n] = vp[d];
}

// ---------------------------------------------------------------- flash attention
// grid = (N/128, B*H); block = 256 (8 waves). K/V tiles staged in LDS via TDM
// (double buffered, s_wait_tensorcnt) or cooperative loads as fallback.
// Softmax scale folded into the Q fragments; normalizer applied via v_rcp.
__global__ void __launch_bounds__(256)
attn_flash(const _Float16* __restrict__ Qh, const _Float16* __restrict__ Kh,
           const _Float16* __restrict__ Vt, _Float16* __restrict__ AO) {
  extern __shared__ _Float16 smem[];
  _Float16* smemK = smem;                            // 2 * KTILE_H
  _Float16* smemV = smem + 2 * KTILE_H;              // 2 * VTILE_H
  _Float16* smemP = smem + 2 * KTILE_H + 2 * VTILE_H;

  const int lane = threadIdx.x & 31;
  const int wave = threadIdx.x >> 5;
  const int lhi  = lane >> 4;
  const int lsub = lane & 15;
  const int bh = blockIdx.y;
  const int b = bh >> 3, h = bh & 7;
  const int q0 = blockIdx.x * 128 + wave * 16;

  const _Float16* Qb = Qh + (size_t)bh * N_ * DH_;
  const _Float16* Kb = Kh + (size_t)bh * N_ * DH_;
  const _Float16* Vb = Vt + (size_t)bh * DH_ * N_;
  _Float16* Pw = smemP + wave * 16 * PSTR;
  const float scale = 0.04419417382415922f;  // 512^-0.5 (reference scales by full D)

  // Q fragments with the softmax scale pre-folded: S*scale == (Q*scale)*K^T
  v16h qf[2];
  {
    const size_t qrow = (size_t)(q0 + lsub) * DH_;
    qf[0] = *(const v16h*)(Qb + qrow + 16 * lhi);
    qf[1] = *(const v16h*)(Qb + qrow + 32 + 16 * lhi);
    const _Float16 hs = (_Float16)scale;
#pragma unroll
    for (int i = 0; i < 16; ++i) { qf[0][i] *= hs; qf[1][i] *= hs; }
  }

  float mrow[8], lsum[8];
#pragma unroll
  for (int r = 0; r < 8; ++r) { mrow[r] = -1e30f; lsum[r] = 0.f; }
  v8f o[4] = {};

  const int NBLK = N_ / 128;
#if USE_TDM
  if (wave == 0) {
    // prologue: stage block 0 into buffer 0
    tdm_load_2d((unsigned)(uintptr_t)(smemK), Kb, 64, 128, 64, N_, DH_, 4, 3);
    tdm_load_2d((unsigned)(uintptr_t)(smemV), Vb, 128, 64, N_, DH_, N_, 5, 3);
  }
#endif

  for (int ib = 0; ib < NBLK; ++ib) {
    const int kb = ib * 128;
    const int buf = ib & 1;
    _Float16* Kt = smemK + buf * KTILE_H;
    _Float16* Vl = smemV + buf * VTILE_H;

#if USE_TDM
    if (wave == 0) {
      if (ib + 1 < NBLK) {
        _Float16* Kn = smemK + (buf ^ 1) * KTILE_H;
        _Float16* Vn = smemV + (buf ^ 1) * VTILE_H;
        tdm_load_2d((unsigned)(uintptr_t)Kn, Kb + (size_t)(kb + 128) * DH_,
                    64, 128, 64, N_, DH_, 4, 3);
        tdm_load_2d((unsigned)(uintptr_t)Vn, Vb + (kb + 128),
                    128, 64, N_, DH_, N_, 5, 3);
        __builtin_amdgcn_s_wait_tensorcnt(2);  // current buffer's 2 DMAs done
      } else {
        __builtin_amdgcn_s_wait_tensorcnt(0);
      }
    }
    __syncthreads();
#else
    // cooperative staging fallback (padded rows, 16B chunks)
    for (int t = threadIdx.x; t < 128 * 8; t += 256) {
      const int row = t >> 3, c = (t & 7) * 8;
      *(v8h*)(Kt + row * KSTR + c) = *(const v8h*)(Kb + (size_t)(kb + row) * DH_ + c);
    }
    for (int t = threadIdx.x; t < 64 * 16; t += 256) {
      const int d = t >> 4, c = (t & 15) * 8;
      *(v8h*)(Vl + d * VSTR + c) = *(const v8h*)(Vb + (size_t)d * N_ + kb + c);
    }
    __syncthreads();
#endif

    // ---- S = (Q*scale) * K^T (16x128): fragment batches, then WMMA bursts
    v8f s[8] = {};
    {
      v16h kf[8];
#pragma unroll
      for (int nc = 0; nc < 8; ++nc)
        kf[nc] = *(const v16h*)(Kt + (nc * 16 + lsub) * KSTR + 16 * lhi);
#pragma unroll
      for (int nc = 0; nc < 8; ++nc) s[nc] = wmma_f16f32(qf[0], kf[nc], s[nc]);
#pragma unroll
      for (int nc = 0; nc < 8; ++nc)
        kf[nc] = *(const v16h*)(Kt + (nc * 16 + lsub) * KSTR + 32 + 16 * lhi);
#pragma unroll
      for (int nc = 0; nc < 8; ++nc) s[nc] = wmma_f16f32(qf[1], kf[nc], s[nc]);
    }

    // ---- online softmax over this key block
    float al[8];
#pragma unroll
    for (int r = 0; r < 8; ++r) {
      float v = -1e30f;
#pragma unroll
      for (int nc = 0; nc < 8; ++nc) v = fmaxf(v, s[nc][r]);
      v = fmaxf(v, __shfl_xor(v, 1, 32));
      v = fmaxf(v, __shfl_xor(v, 2, 32));
      v = fmaxf(v, __shfl_xor(v, 4, 32));
      v = fmaxf(v, __shfl_xor(v, 8, 32));
      const float mnew  = fmaxf(mrow[r], v);
      const float alpha = __expf(mrow[r] - mnew);
      float rs = 0.f;
#pragma unroll
      for (int nc = 0; nc < 8; ++nc) {
        const float p = __expf(s[nc][r] - mnew);
        rs += p;
        Pw[(r + 8 * lhi) * PSTR + nc * 16 + lsub] = (_Float16)p;
      }
      rs += __shfl_xor(rs, 1, 32); rs += __shfl_xor(rs, 2, 32);
      rs += __shfl_xor(rs, 4, 32); rs += __shfl_xor(rs, 8, 32);
      lsum[r] = lsum[r] * alpha + rs;
      mrow[r] = mnew;
      al[r] = alpha;
    }
#pragma unroll
    for (int nc = 0; nc < 4; ++nc)
#pragma unroll
      for (int r = 0; r < 8; ++r) o[nc][r] *= al[r];
    __syncthreads();

    // ---- O += P * V : P re-read as A-fragments, V fragments preloaded per kc
#pragma unroll
    for (int kc = 0; kc < 4; ++kc) {
      const _Float16* pr = Pw + lsub * PSTR;
      v16h pf = *(const v16h*)(pr + kc * 32 + 16 * lhi);
      v16h vf[4];
#pragma unroll
      for (int nc = 0; nc < 4; ++nc)
        vf[nc] = *(const v16h*)(Vl + (nc * 16 + lsub) * VSTR + kc * 32 + 16 * lhi);
#pragma unroll
      for (int nc = 0; nc < 4; ++nc) o[nc] = wmma_f16f32(pf, vf[nc], o[nc]);
    }
    __syncthreads();
  }

  // ---- epilogue: O * (1/l) via v_rcp, written in (B, N, H*Dh) layout
  float inv[8];
#pragma unroll
  for (int r = 0; r < 8; ++r) inv[r] = __builtin_amdgcn_rcpf(lsum[r]);
#pragma unroll
  for (int nc = 0; nc < 4; ++nc) {
#pragma unroll
    for (int r = 0; r < 8; ++r) {
      const int qr = q0 + r + 8 * lhi;
      const float v = o[nc][r] * inv[r];
      AO[((size_t)(b * N_) + qr) * D_ + h * DH_ + nc * 16 + lsub] = (_Float16)v;
    }
  }
}

// ---------------------------------------------------------------- launch
extern "C" void kernel_launch(void* const* d_in, const int* in_sizes, int n_in,
                              void* d_out, int out_size, void* d_ws, size_t ws_size,
                              hipStream_t stream) {
  (void)in_sizes; (void)n_in; (void)out_size; (void)ws_size;
  const float* x     = (const float*)d_in[0];
  const float* pos   = (const float*)d_in[1];
  const float* qkv_w = (const float*)d_in[2];
  const float* qkv_b = (const float*)d_in[3];
  const float* out_w = (const float*)d_in[4];
  const float* out_b = (const float*)d_in[5];
  float* out = (float*)d_out;

  char* p = (char*)d_ws;
  _Float16* xh    = (_Float16*)p; p += (size_t)B_ * N_ * D_ * 2;
  _Float16* wqkvh = (_Float16*)p; p += (size_t)TD_ * D_ * 2;
  _Float16* wouth = (_Float16*)p; p += (size_t)D_ * D_ * 2;
  _Float16* qkvh  = (_Float16*)p; p += (size_t)B_ * N_ * TD_ * 2;
  _Float16* Qh    = (_Float16*)p; p += (size_t)B_ * H_ * N_ * DH_ * 2;
  _Float16* Kh    = (_Float16*)p; p += (size_t)B_ * H_ * N_ * DH_ * 2;
  _Float16* Vt    = (_Float16*)p; p += (size_t)B_ * H_ * DH_ * N_ * 2;
  _Float16* ao    = (_Float16*)p; p += (size_t)B_ * N_ * D_ * 2;

  cast_f32_to_f16<<<1024, 256, 0, stream>>>(x, xh, B_ * N_ * D_);
  cast_f32_to_f16<<<512, 256, 0, stream>>>(qkv_w, wqkvh, TD_ * D_);
  cast_f32_to_f16<<<256, 256, 0, stream>>>(out_w, wouth, D_ * D_);

  gemm_xwT<TD_, true><<<dim3(64, 24), 256, 0, stream>>>(xh, wqkvh, qkv_b, qkvh, D_);

  rot_split<<<(B_ * N_ * H_) / 256, 256, 0, stream>>>(qkvh, pos, Qh, Kh, Vt);

  attn_flash<<<dim3(N_ / 128, B_ * H_), 256, SMEM_BYTES, stream>>>(Qh, Kh, Vt, ao);

  gemm_xwT<D_, false><<<dim3(64, 8), 256, 0, stream>>>(ao, wouth, out_b, out, D_);
}
